// InterpretableMultiHeadAttention_6828998001104
// MI455X (gfx1250) — compile-verified, matrix operand order checked by reading, not run on hardware
//
#include <hip/hip_runtime.h>
#include <cstdint>
#include <cstddef>

// ---------------------------------------------------------------------------
// InterpretableMultiHeadAttention for MI455X (gfx1250, wave32, WMMA)
//   BS=4, T=2048, H=512, NH=8, DH=64; V shared across heads.
//   d_out = [out (4*2048*512 f32)] ++ [attn_prob (4*8*2048*2048 f32)]
// Big GEMMs: v_wmma_f32_16x16x32_f16 (f16 in, f32 acc).
// Final projection: v_wmma_f32_16x16x4_f32 (full fp32).
// Softmax: sweep1 = row max only (no transcendentals in hot loop),
//          sweep2 = single exp2 per element + free row-sum, then an
//          L2-hot normalize fix-up pass over the wave's own rows.
// ---------------------------------------------------------------------------

typedef __attribute__((ext_vector_type(16))) _Float16 v16h;
typedef __attribute__((ext_vector_type(8)))  float    v8f;
typedef __attribute__((ext_vector_type(2)))  float    v2f;
typedef __attribute__((ext_vector_type(4)))  float    f32x4;
typedef __attribute__((ext_vector_type(4)))  unsigned int u32x4;

#define BS   4
#define T    2048
#define H    512
#define NH   8
#define DH   64
#define NQKV 1088                     // (2*NH+1)*DH
#define QSCALE 0.18033688011112042f   // (1/8) * log2(e): softmax done in exp2 domain

union H16 { u32x4 u[2]; v16h h; };

static __device__ __forceinline__ v16h load_h16(const _Float16* p0, const _Float16* p1) {
  H16 t;
  t.u[0] = *(const u32x4*)p0;   // halves 0..7   (16B aligned by construction)
  t.u[1] = *(const u32x4*)p1;   // halves 8..15
  return t.h;
}

// ---------------- K0: w_qkv [512 x 1088] f32 -> wT [1088 x 512] f16 ----------
__global__ void k_wt(const float* __restrict__ w, _Float16* __restrict__ wT) {
  int idx = blockIdx.x * 256 + threadIdx.x;
  if (idx >= NQKV * H) return;
  int n = idx / H, k = idx % H;        // wT[n][k] = w[k][n]
  wT[idx] = (_Float16)w[(size_t)k * NQKV + n];
}

// ---------------- K1: qkv = x @ w_qkv -> Qh (scaled), Kh, Vt ----------------
// One wave computes one 16x16 tile of [8192 x 1088]. K loop = 16 WMMAs.
__global__ void __launch_bounds__(128)
k_qkv(const float* __restrict__ x, const _Float16* __restrict__ wT,
      _Float16* __restrict__ Qh, _Float16* __restrict__ Kh,
      _Float16* __restrict__ Vt) {
  const int wid  = (blockIdx.x * blockDim.x + threadIdx.x) >> 5;
  const int lane = threadIdx.x & 31;
  const int ln = lane & 15, hi = lane >> 4;
  const int NT = NQKV / 16;            // 68 N-tiles
  int mt = wid / NT, nt = wid % NT;
  if (mt >= (BS * T) / 16) return;
  int n0 = nt * 16;

  v8f acc = {};
  for (int k0 = 0; k0 < H; k0 += 32) {
    int c0 = k0 + (hi ? 8 : 0);
    // A fragment: x rows (fp32 -> f16), 16-bit A 16x32 layout
    const float* xr = x + (size_t)(mt * 16 + ln) * H + c0;
    f32x4 x0 = *(const f32x4*)(xr);
    f32x4 x1 = *(const f32x4*)(xr + 4);
    f32x4 x2 = *(const f32x4*)(xr + 16);
    f32x4 x3 = *(const f32x4*)(xr + 20);
    v16h a;
#pragma unroll
    for (int i = 0; i < 4; ++i) {
      a[i]      = (_Float16)x0[i];
      a[4 + i]  = (_Float16)x1[i];
      a[8 + i]  = (_Float16)x2[i];
      a[12 + i] = (_Float16)x3[i];
    }
    // B fragment: wT row n = output column, contiguous k
    const _Float16* wr = wT + (size_t)(n0 + ln) * H + c0;
    v16h b = load_h16(wr, wr + 16);
    acc = __builtin_amdgcn_wmma_f32_16x16x32_f16(false, a, false, b,
                                                 (short)0, acc, false, false);
  }

  int colbase = n0 + ln;               // tile never straddles Q/K/V or head bounds
#pragma unroll
  for (int r = 0; r < 8; ++r) {
    int m = mt * 16 + r + 8 * hi;
    int b_ = m / T, t = m % T;
    float v = acc[r];
    if (n0 < H) {                                        // Q (pre-scaled)
      int head = colbase / DH, d = colbase % DH;
      Qh[(((size_t)(b_ * NH + head)) * T + t) * DH + d] = (_Float16)(v * QSCALE);
    } else if (n0 < 2 * H) {                             // K
      int c2 = colbase - H;
      int head = c2 / DH, d = c2 % DH;
      Kh[(((size_t)(b_ * NH + head)) * T + t) * DH + d] = (_Float16)v;
    } else {                                             // V -> transposed [64 x T]
      int d = colbase - 2 * H;
      Vt[((size_t)b_ * DH + d) * T + t] = (_Float16)v;
    }
  }
}

// ---------------- K2: causal attention, prob writeout, P@V ------------------
// One wave per (b, h, 16-row q-tile).
//   sweep 1: row max via QK^T WMMA (fmax only) + shfl_xor reduce
//   sweep 2: recompute S, p=exp2(s-m) once, l += p, store raw p to attn_prob,
//            LDS-bounce p->f16 A-frags, accumulate O via WMMA against Vt.
//   epilog : O *= 1/l; L2-hot fix-up pass rescales this wave's attn_prob rows;
//            zero-fill strictly-future columns.
__global__ void __launch_bounds__(128)
k_attn(const _Float16* __restrict__ Qh, const _Float16* __restrict__ Kh,
       const _Float16* __restrict__ Vt, float* __restrict__ P,
       float* __restrict__ av) {
  __shared__ _Float16 stage[4][16 * 32];     // 1KB per wave P re-layout tile
  const int w    = threadIdx.x >> 5;
  const int lane = threadIdx.x & 31;
  const int ln = lane & 15, hi = lane >> 4;
  int gid = blockIdx.x * 4 + w;
  const int NQT = T / 16;                    // 128
  int qt = gid % NQT;
  int bh = gid / NQT;
  int b  = bh / NH;
  if (b >= BS) return;

  const _Float16* Qb = Qh + (size_t)bh * T * DH;
  const _Float16* Kb = Kh + (size_t)bh * T * DH;
  const _Float16* Vb = Vt + (size_t)b * DH * T;
  float* Pb = P + (size_t)bh * T * T;

  const int c0 = hi ? 8 : 0;
  // Q A-fragments (d=64 -> two 16x32 frags), loaded once
  const _Float16* qr = Qb + (size_t)(qt * 16 + ln) * DH;
  v16h aq0 = load_h16(qr + c0,      qr + c0 + 16);
  v16h aq1 = load_h16(qr + 32 + c0, qr + 32 + c0 + 16);

  // ---- sweep 1: row max only ----
  float m[8];
#pragma unroll
  for (int r = 0; r < 8; ++r) m[r] = -1e30f;

  for (int kt = 0; kt <= qt; ++kt) {
    const _Float16* kr = Kb + (size_t)(kt * 16 + ln) * DH;
    v16h bk0 = load_h16(kr + c0,      kr + c0 + 16);
    v16h bk1 = load_h16(kr + 32 + c0, kr + 32 + c0 + 16);
    if (kt < qt)
      __builtin_prefetch(Kb + (size_t)((kt + 1) * 16 + ln) * DH, 0, 0);
    v8f s = {};
    s = __builtin_amdgcn_wmma_f32_16x16x32_f16(false, aq0, false, bk0, (short)0, s, false, false);
    s = __builtin_amdgcn_wmma_f32_16x16x32_f16(false, aq1, false, bk1, (short)0, s, false, false);
    int col = kt * 16 + ln;
#pragma unroll
    for (int r = 0; r < 8; ++r) {
      float sv = s[r];
      if (kt == qt && col > qt * 16 + r + 8 * hi) sv = -__builtin_inff();
      m[r] = fmaxf(m[r], sv);
    }
  }
#pragma unroll
  for (int msk = 1; msk < 16; msk <<= 1)
#pragma unroll
    for (int r = 0; r < 8; ++r)
      m[r] = fmaxf(m[r], __shfl_xor(m[r], msk, 32));

  // ---- sweep 2: p once, row sum for free, raw p out, O accumulation ----
  float l[8];
#pragma unroll
  for (int r = 0; r < 8; ++r) l[r] = 0.f;
  v8f o0 = {}, o1 = {}, o2 = {}, o3 = {};
  _Float16* st = stage[w];

  for (int kp = 0; kp <= qt / 2; ++kp) {
#pragma unroll
    for (int half = 0; half < 2; ++half) {
      int kt = kp * 2 + half;
      if (kt <= qt) {
        const _Float16* kr = Kb + (size_t)(kt * 16 + ln) * DH;
        v16h bk0 = load_h16(kr + c0,      kr + c0 + 16);
        v16h bk1 = load_h16(kr + 32 + c0, kr + 32 + c0 + 16);
        v8f s = {};
        s = __builtin_amdgcn_wmma_f32_16x16x32_f16(false, aq0, false, bk0, (short)0, s, false, false);
        s = __builtin_amdgcn_wmma_f32_16x16x32_f16(false, aq1, false, bk1, (short)0, s, false, false);
        int col = kt * 16 + ln;
#pragma unroll
        for (int r = 0; r < 8; ++r) {
          float sv = s[r];
          int row = qt * 16 + r + 8 * hi;
          if (kt == qt && col > row) sv = -__builtin_inff();
          float p = exp2f(sv - m[r]);                 // masked -> exactly 0
          l[r] += p;
          Pb[(size_t)row * T + col] = p;              // raw; normalized in fix-up
          st[(r + 8 * hi) * 32 + half * 16 + ln] = (_Float16)p;
        }
      } else {
        // final incomplete pair: zero upper 16 columns of the stage tile
        u32x4 z = {};
        *(u32x4*)(st + ln * 32 + 16 + hi * 8) = z;
      }
    }
    // wave-coherent LDS: drain DS stores before re-layout loads
    asm volatile("s_wait_dscnt 0" ::: "memory");
    v16h pa = load_h16(st + ln * 32 + c0, st + ln * 32 + c0 + 16);
    int t0 = kp * 32 + c0;
    const _Float16* v0 = Vb + (size_t)(0 * 16 + ln) * T + t0;
    const _Float16* v1 = Vb + (size_t)(1 * 16 + ln) * T + t0;
    const _Float16* v2 = Vb + (size_t)(2 * 16 + ln) * T + t0;
    const _Float16* v3 = Vb + (size_t)(3 * 16 + ln) * T + t0;
    o0 = __builtin_amdgcn_wmma_f32_16x16x32_f16(false, pa, false, load_h16(v0, v0 + 16), (short)0, o0, false, false);
    o1 = __builtin_amdgcn_wmma_f32_16x16x32_f16(false, pa, false, load_h16(v1, v1 + 16), (short)0, o1, false, false);
    o2 = __builtin_amdgcn_wmma_f32_16x16x32_f16(false, pa, false, load_h16(v2, v2 + 16), (short)0, o2, false, false);
    o3 = __builtin_amdgcn_wmma_f32_16x16x32_f16(false, pa, false, load_h16(v3, v3 + 16), (short)0, o3, false, false);
  }

  // reduce row sums across the 16 lanes holding each row (plain adds)
#pragma unroll
  for (int msk = 1; msk < 16; msk <<= 1)
#pragma unroll
    for (int r = 0; r < 8; ++r)
      l[r] += __shfl_xor(l[r], msk, 32);
  float invl[8];
#pragma unroll
  for (int r = 0; r < 8; ++r) invl[r] = 1.0f / l[r];

  // attn_vec (normalized) -> workspace, f32, deterministic (no atomics)
#pragma unroll
  for (int r = 0; r < 8; ++r) {
    size_t rowoff = ((size_t)bh * T + qt * 16 + r + 8 * hi) * DH;
    av[rowoff + 0 * 16 + ln] = o0[r] * invl[r];
    av[rowoff + 1 * 16 + ln] = o1[r] * invl[r];
    av[rowoff + 2 * 16 + ln] = o2[r] * invl[r];
    av[rowoff + 3 * 16 + ln] = o3[r] * invl[r];
  }

  // fix-up: normalize this wave's attn_prob rows (L2-hot, same-wave ordering)
  int ncol = (qt + 1) * 16;
#pragma unroll 1
  for (int r16 = 0; r16 < 16; ++r16) {
    float sc = __shfl(invl[r16 & 7], (r16 >> 3) * 16, 32);
    float* prow = Pb + (size_t)(qt * 16 + r16) * T;
    for (int col = lane * 4; col < ncol; col += 128) {
      f32x4 v = *(const f32x4*)(prow + col);
      v = v * sc;
      *(f32x4*)(prow + col) = v;
    }
  }

  // zero-fill masked (strictly-future) columns of attn_prob
  if (ncol < T) {
    f32x4 z = {};
#pragma unroll 1
    for (int r16 = 0; r16 < 16; ++r16) {
      float* prow = Pb + (size_t)(qt * 16 + r16) * T;
      for (int col = ncol + lane * 4; col < T; col += 128)
        *(f32x4*)(prow + col) = z;
    }
  }
}

// ---------------- K3a: mean over heads ------------------------------------
__global__ void k_mean(const float* __restrict__ av, float* __restrict__ mvec) {
  int idx = blockIdx.x * 256 + threadIdx.x;
  if (idx >= BS * T * DH) return;
  int d = idx % DH;
  int t = (idx / DH) % T;
  int b = idx / (DH * T);
  float s = 0.f;
#pragma unroll
  for (int h = 0; h < NH; ++h)
    s += av[(((size_t)(b * NH + h)) * T + t) * DH + d];
  mvec[idx] = s * (1.0f / NH);
}

// ---------------- K3b: out = mvec @ w_out (full fp32 WMMA) ------------------
__global__ void __launch_bounds__(128)
k_out(const float* __restrict__ mvec, const float* __restrict__ Wo,
      float* __restrict__ out) {
  const int wid  = (blockIdx.x * blockDim.x + threadIdx.x) >> 5;
  const int lane = threadIdx.x & 31;
  const int ln = lane & 15, hi = lane >> 4;
  const int NT = H / 16;               // 32
  int mt = wid / NT, nt = wid % NT;
  if (mt >= (BS * T) / 16) return;
  int n = nt * 16 + ln;

  v8f acc = {};
  const float* arow = mvec + (size_t)(mt * 16 + ln) * DH;
#pragma unroll
  for (int kk = 0; kk < DH; kk += 4) {
    int k = kk + hi * 2;               // 32-bit A 16x4: lanes 16-31 hold K=2,3
    v2f a = *(const v2f*)(arow + k);
    v2f bb;
    bb[0] = Wo[(size_t)k * H + n];
    bb[1] = Wo[(size_t)(k + 1) * H + n];
    acc = __builtin_amdgcn_wmma_f32_16x16x4_f32(false, a, false, bb,
                                                (short)0, acc, false, false);
  }
#pragma unroll
  for (int r = 0; r < 8; ++r)
    out[(size_t)(mt * 16 + r + 8 * hi) * H + nt * 16 + ln] = acc[r];
}

// ---------------------------------------------------------------------------
extern "C" void kernel_launch(void* const* d_in, const int* in_sizes, int n_in,
                              void* d_out, int out_size, void* d_ws, size_t ws_size,
                              hipStream_t stream) {
  const float* x     = (const float*)d_in[0];   // [4,2048,512]
  const float* w_qkv = (const float*)d_in[1];   // [512,1088]
  const float* w_out = (const float*)d_in[2];   // [64,512]
  float* out = (float*)d_out;                   // [4,2048,512]
  float* P   = out + (size_t)BS * T * H;        // [4,8,2048,2048]

  // workspace carve-up (~38 MB total, all sections 1KB aligned)
  char* ws = (char*)d_ws;
  _Float16* Qh = (_Float16*)ws;  ws += (size_t)BS * NH * T * DH * 2;   // 8 MB
  _Float16* Kh = (_Float16*)ws;  ws += (size_t)BS * NH * T * DH * 2;   // 8 MB
  _Float16* Vt = (_Float16*)ws;  ws += (size_t)BS * DH * T * 2;        // 1 MB
  _Float16* wT = (_Float16*)ws;  ws += (size_t)NQKV * H * 2;           // ~1.1 MB
  float*    av = (float*)ws;     ws += (size_t)BS * NH * T * DH * 4;   // 16 MB
  float*  mvec = (float*)ws;                                           // 2 MB

  k_wt<<<(NQKV * H + 255) / 256, 256, 0, stream>>>(w_qkv, wT);

  {
    int tiles = (BS * T / 16) * (NQKV / 16);      // 34816 waves
    k_qkv<<<(tiles + 3) / 4, 128, 0, stream>>>(x, wT, Qh, Kh, Vt);
  }
  {
    int waves = BS * NH * (T / 16);               // 4096 waves
    k_attn<<<waves / 4, 128, 0, stream>>>(Qh, Kh, Vt, P, av);
  }
  k_mean<<<(BS * T * DH + 255) / 256, 256, 0, stream>>>(av, mvec);
  {
    int tiles = (BS * T / 16) * (H / 16);         // 16384 waves
    k_out<<<(tiles + 3) / 4, 128, 0, stream>>>(mvec, w_out, out);
  }
}
